// NDTRouter_360777253222
// MI455X (gfx1250) — matrix operand order repaired and checked by reading, not compiled, not run on hardware
//
#include <hip/hip_runtime.h>
#include <hip/hip_bf16.h>
#include <math.h>

typedef __attribute__((ext_vector_type(16))) __bf16 v16bf;
typedef __attribute__((ext_vector_type(8)))  float  v8f;

#define HSIZE  4096
#define SCAN_T 256
#define CHUNK  (HSIZE / SCAN_T)
#define NPAD   16      // B matrix padded to 16 columns (D=3 live)
#define KSPLIT 4       // waves cooperating on one 16-row tile (K partitioned)

// ---------------------------------------------------------------------------
// Kernel 1: exact sort-based entmax15 over each feature_selectors row.
// grid = NPAD blocks; rows >= D just zero-fill their padded bf16 B row.
// Output: selb (bf16, [NPAD][H]) ready to be consumed as the WMMA B operand.
// ---------------------------------------------------------------------------
__global__ __launch_bounds__(256) void entmax15_rows_kernel(
    const float* __restrict__ fs, __bf16* __restrict__ selb, int H, int D)
{
    const int row = blockIdx.x;
    const int tid = threadIdx.x;

    if (row >= D) {  // block-uniform: pad rows are zero
        for (int i = tid; i < HSIZE; i += SCAN_T) selb[(size_t)row * H + i] = (__bf16)0.f;
        return;
    }

    __shared__ float xs[HSIZE];
    __shared__ float cs1[HSIZE];
    __shared__ float cs2[HSIZE];
    __shared__ float reds[SCAN_T];
    __shared__ float redq[SCAN_T];
    __shared__ float s_rowmax, s_tau;
    __shared__ int   s_support;

    const float* xrow = fs + (size_t)row * H;

    // x = fs/2, row max
    float lmax = -INFINITY;
    for (int i = tid; i < HSIZE; i += SCAN_T) {
        float v = 0.5f * xrow[i];
        xs[i] = v;
        lmax = fmaxf(lmax, v);
    }
    reds[tid] = lmax;
    __syncthreads();
    for (int off = SCAN_T >> 1; off > 0; off >>= 1) {
        if (tid < off) reds[tid] = fmaxf(reds[tid], reds[tid + off]);
        __syncthreads();
    }
    if (tid == 0) s_rowmax = reds[0];
    __syncthreads();
    const float rowmax = s_rowmax;
    for (int i = tid; i < HSIZE; i += SCAN_T) xs[i] -= rowmax;

    // bitonic sort, descending
    for (unsigned kk = 2; kk <= HSIZE; kk <<= 1) {
        for (unsigned j = kk >> 1; j > 0; j >>= 1) {
            __syncthreads();
            for (unsigned i = tid; i < HSIZE; i += SCAN_T) {
                unsigned ixj = i ^ j;
                if (ixj > i) {
                    float a = xs[i], b = xs[ixj];
                    bool desc = ((i & kk) == 0);
                    if (desc ? (a < b) : (a > b)) { xs[i] = b; xs[ixj] = a; }
                }
            }
        }
    }
    __syncthreads();

    // inclusive prefix sums of xs and xs^2 (chunked scan)
    {
        const int base = tid * CHUNK;
        float s = 0.f, q = 0.f;
        for (int t = 0; t < CHUNK; ++t) {
            float v = xs[base + t];
            s += v;      cs1[base + t] = s;
            q += v * v;  cs2[base + t] = q;
        }
        reds[tid] = s; redq[tid] = q;
    }
    __syncthreads();
    if (tid == 0) {
        float a = 0.f, b = 0.f;
        for (int t = 0; t < SCAN_T; ++t) {
            float ts = reds[t], tq = redq[t];
            reds[t] = a; redq[t] = b;
            a += ts; b += tq;
        }
        s_support = 0;
    }
    __syncthreads();
    {
        const int base = tid * CHUNK;
        float o1 = reds[tid], o2 = redq[tid];
        for (int t = 0; t < CHUNK; ++t) { cs1[base + t] += o1; cs2[base + t] += o2; }
    }
    __syncthreads();

    // support size = #(tau_k <= xs_k)
    int cnt = 0;
    for (int i = tid; i < HSIZE; i += SCAN_T) {
        float k    = (float)(i + 1);
        float mean = cs1[i] / k;
        float msq  = cs2[i] / k;
        float ss   = k * (msq - mean * mean);
        float dl   = (1.f - ss) / k;
        float tau  = mean - sqrtf(fmaxf(dl, 0.f));
        cnt += (tau <= xs[i]) ? 1 : 0;
    }
    atomicAdd(&s_support, cnt);
    __syncthreads();
    if (tid == 0) {
        int sup = s_support < 1 ? 1 : s_support;
        float k    = (float)sup;
        float mean = cs1[sup - 1] / k;
        float msq  = cs2[sup - 1] / k;
        float ss   = k * (msq - mean * mean);
        float dl   = (1.f - ss) / k;
        s_tau = mean - sqrtf(fmaxf(dl, 0.f));
    }
    __syncthreads();
    const float tau_star = s_tau;
    for (int i = tid; i < HSIZE; i += SCAN_T) {
        float x = 0.5f * xrow[i] - rowmax;
        float v = fmaxf(x - tau_star, 0.f);
        selb[(size_t)row * H + i] = (__bf16)(v * v);
    }
}

// ---------------------------------------------------------------------------
// Kernel 2: fused WMMA skinny-GEMM + tree-routing epilogue.
// Each 16-row tile is computed by KSPLIT=4 cooperating waves (K partitioned)
// to reach the HBM bandwidth-delay product; partials are summed through LDS.
// Block = 256 threads = 8 waves = 2 tiles.
// ---------------------------------------------------------------------------
__global__ __launch_bounds__(256) void ndt_router_wmma_kernel(
    const float* __restrict__ hs,  const __bf16* __restrict__ selb,
    const float* __restrict__ thr, const float* __restrict__ logt,
    const float* __restrict__ Wl,  const float* __restrict__ bl,
    float* __restrict__ out_logits, int* __restrict__ out_idx,
    float* __restrict__ out_w, int H, int rows, int tiles)
{
    __shared__ float partLds[2][KSPLIT][16][16];   // 8 KB

    const int lane      = threadIdx.x & 31;
    const int wave      = threadIdx.x >> 5;
    const int tileInBlk = wave >> 2;               // 0..1
    const int ks        = wave & (KSPLIT - 1);     // 0..3 K-slice
    const int tileIdx   = blockIdx.x * 2 + tileInBlk;
    const bool active   = (tileIdx < tiles);       // wave-uniform
    const int r0   = tileIdx * 16;
    const int half = lane >> 4;                    // K-half select (wave32 layout)
    const int mrow = lane & 15;                    // A row this lane feeds

    v8f acc = {0.f, 0.f, 0.f, 0.f, 0.f, 0.f, 0.f, 0.f};

    if (active) {
        const int kslice = H / KSPLIT;             // 1024
        const int kbeg = ks * kslice;
        const int kend = kbeg + kslice;

        // A: row r0+mrow; lane half picks K groups {0..15} / {16..31}
        const float* arow = hs + (size_t)(r0 + mrow) * H + half * 16;
        // B: column n reads padded bf16 row n; 16 contiguous bf16 per lane
        const __bf16* brow = selb + (size_t)(lane & 15) * H + half * 16;

        #pragma unroll 2
        for (int k0 = kbeg; k0 < kend; k0 += 32) {
            const float4* ag = (const float4*)(arow + k0);
            float4 a0 = ag[0], a1 = ag[1];   // K lo group (8 floats)
            float4 a2 = ag[2], a3 = ag[3];   // K next group (+8 floats)
            v16bf B = *(const v16bf*)(brow + k0);   // 32 B -> 2x b128
            __builtin_prefetch(arow + k0 + 512, 0, 0);

            v16bf A;
            A[0]  = (__bf16)a0.x; A[1]  = (__bf16)a0.y; A[2]  = (__bf16)a0.z; A[3]  = (__bf16)a0.w;
            A[4]  = (__bf16)a1.x; A[5]  = (__bf16)a1.y; A[6]  = (__bf16)a1.z; A[7]  = (__bf16)a1.w;
            A[8]  = (__bf16)a2.x; A[9]  = (__bf16)a2.y; A[10] = (__bf16)a2.z; A[11] = (__bf16)a2.w;
            A[12] = (__bf16)a3.x; A[13] = (__bf16)a3.y; A[14] = (__bf16)a3.z; A[15] = (__bf16)a3.w;

            acc = __builtin_amdgcn_wmma_f32_16x16x32_bf16(
                false, A, false, B, (short)0, acc, false, false);
        }

        // C layout: VGPR v -> M = v + 8*(lane>=16), N = lane&15
        #pragma unroll
        for (int v = 0; v < 8; ++v)
            partLds[tileInBlk][ks][v + half * 8][lane & 15] = acc[v];
    }
    __syncthreads();

    if (active && ks == 0 && lane < 16) {
        const int r = r0 + lane;
        float fc[3];
        #pragma unroll
        for (int d = 0; d < 3; ++d) {
            float s = 0.f;
            #pragma unroll
            for (int q = 0; q < KSPLIT; ++q) s += partLds[tileInBlk][q][lane][d];
            fc[d] = s;
        }
        float right[3], left[3];
        #pragma unroll
        for (int d = 0; d < 3; ++d) {
            float s = (fc[d] - thr[d]) * __expf(-logt[d]);
            float u = 0.5f * fabsf(s);
            float pt, pb;
            if (u <= 1.f) {
                float dl = 0.5f * (1.f - 0.5f * u * u);
                float t  = -0.5f * u - sqrtf(fmaxf(dl, 0.f));
                pt = t * t;
                float d2 = fmaxf(-u - t, 0.f);
                pb = d2 * d2;
            } else { pt = 1.f; pb = 0.f; }
            float rr = (s > 0.f) ? pt : pb;
            right[d] = rr; left[d] = 1.f - rr;
        }
        // depth-3 oblivious tree leaf probabilities
        float leaf[8];
        #pragma unroll
        for (int mm = 0; mm < 8; ++mm) {
            float p = ((mm >> 2) & 1) ? right[0] : left[0];
            p      *= ((mm >> 1) & 1) ? right[1] : left[1];
            p      *= (mm & 1)        ? right[2] : left[2];
            leaf[mm] = p;
        }
        // expert logits: 8x8 matvec + bias
        float lg[8];
        #pragma unroll
        for (int e = 0; e < 8; ++e) {
            float s = bl[e];
            #pragma unroll
            for (int mm = 0; mm < 8; ++mm) s += leaf[mm] * Wl[e * 8 + mm];
            lg[e] = s;
        }
        // top-2 (ties -> lower index, matching lax.top_k) + softmax over the pair
        int e1 = 0; float l1 = lg[0];
        #pragma unroll
        for (int e = 1; e < 8; ++e) if (lg[e] > l1) { l1 = lg[e]; e1 = e; }
        int e2 = 0; float l2 = -INFINITY;
        #pragma unroll
        for (int e = 0; e < 8; ++e) if (e != e1 && lg[e] > l2) { l2 = lg[e]; e2 = e; }
        float w1 = 1.f / (1.f + __expf(l2 - l1));

        #pragma unroll
        for (int e = 0; e < 8; ++e) out_logits[(size_t)r * 8 + e] = lg[e];
        out_idx[(size_t)r * 2 + 0] = e1;
        out_idx[(size_t)r * 2 + 1] = e2;
        out_w[(size_t)r * 2 + 0] = w1;
        out_w[(size_t)r * 2 + 1] = 1.f - w1;
    }
}

// ---------------------------------------------------------------------------
extern "C" void kernel_launch(void* const* d_in, const int* in_sizes, int n_in,
                              void* d_out, int out_size, void* d_ws, size_t ws_size,
                              hipStream_t stream) {
    const float* hs   = (const float*)d_in[0];  // (B,L,H) fp32
    const float* fs   = (const float*)d_in[1];  // (D,H)
    const float* thr  = (const float*)d_in[2];  // (D,1)
    const float* logt = (const float*)d_in[3];  // (D,1)
    const float* Wl   = (const float*)d_in[4];  // (E,2^D)
    const float* bl   = (const float*)d_in[5];  // (E,)

    const int D    = in_sizes[2];                    // 3
    const int H    = in_sizes[1] / (D > 0 ? D : 1);  // 4096
    const int rows = in_sizes[0] / (H > 0 ? H : 1);  // B*L = 16384
    const int E    = in_sizes[5];                    // 8

    __bf16* selb = (__bf16*)d_ws;                    // padded (NPAD,H) bf16 B operand

    float* out_logits = (float*)d_out;                                   // rows*E
    int*   out_idx    = (int*)(out_logits + (size_t)rows * E);           // rows*2 (int32 bits)
    float* out_w      = (float*)(out_logits + (size_t)rows * E + (size_t)rows * 2);

    entmax15_rows_kernel<<<NPAD, 256, 0, stream>>>(fs, selb, H, D);

    const int tiles  = rows / 16;
    const int blocks = (tiles + 1) / 2;              // 2 tiles x 4 K-slice waves per block
    ndt_router_wmma_kernel<<<blocks, 256, 0, stream>>>(
        hs, selb, thr, logt, Wl, bl, out_logits, out_idx, out_w, H, rows, tiles);
}
